// Attention_43963285242501
// MI455X (gfx1250) — matrix operand order repaired and checked by reading, not compiled, not run on hardware
//
#include <hip/hip_runtime.h>
#include <hip/hip_bf16.h>

// ---------------------------------------------------------------------------
// MI455X (gfx1250) fused attention block.
// GEMMs + attention matmuls -> v_wmma_f32_16x16x32_bf16 (wave32 WMMA).
// Global->LDS staging -> global_load_async_to_lds_b128 (ASYNCcnt) when the
// toolchain exposes the builtin, else pipelined b128 register copies.
// ---------------------------------------------------------------------------

typedef unsigned short u16;
typedef unsigned int   u32;

#define B_   2
#define L_   2048
#define D_   2048
#define H_   32
#define HKV_ 4
#define HD_  64
#define M_   (B_ * L_)      // 4096 token rows
#define NQ_  (H_ * HD_)     // 2048
#define NKV_ (HKV_ * HD_)   // 256

typedef __attribute__((ext_vector_type(16))) __bf16 v16bf;
typedef __attribute__((ext_vector_type(8)))  float  v8f;
typedef __attribute__((ext_vector_type(4)))  u32    u32x4;
typedef __attribute__((ext_vector_type(4)))  int    v4i;

#define AS1 __attribute__((address_space(1)))
#define AS3 __attribute__((address_space(3)))

union Frag {
    v16bf v;
    u32   u[8];
};

__device__ __forceinline__ u16 f2bf(float f) {
    u32 x = __float_as_uint(f);
    u32 r = (x + 0x7FFFu + ((x >> 16) & 1u)) >> 16;   // round-to-nearest-even
    return (u16)r;
}

__device__ __forceinline__ v8f v8f_zero() {
    v8f z;
#pragma unroll
    for (int i = 0; i < 8; ++i) z[i] = 0.0f;
    return z;
}

// ---- CDNA5 async global->LDS copy (16 bytes), guarded by __has_builtin -----
// Probe-discovered signature: (v4i addrspace(1)*, v4i addrspace(3)*, imm, imm)
#if __has_builtin(__builtin_amdgcn_global_load_async_to_lds_b128)
#define ASYNC_COPY 1
__device__ __forceinline__ void g2l_b128(const u16* g, u16* l) {
    __builtin_amdgcn_global_load_async_to_lds_b128(
        (AS1 v4i*)(void*)g, (AS3 v4i*)l, 0, 0);
}
#else
#define ASYNC_COPY 0
#endif

__device__ __forceinline__ void async_wait0() {
#if ASYNC_COPY
#if __has_builtin(__builtin_amdgcn_s_wait_asynccnt)
    __builtin_amdgcn_s_wait_asynccnt(0);
#else
    asm volatile("s_wait_asynccnt 0x0" ::: "memory");
#endif
#endif
}

// ---------------------------------------------------------------------------
// fp32 -> bf16 elementwise conversion
// ---------------------------------------------------------------------------
__global__ void cvt_f32_bf16(const float* __restrict__ in, u16* __restrict__ out, size_t n) {
    size_t stride = (size_t)gridDim.x * blockDim.x;
    for (size_t i = (size_t)blockIdx.x * blockDim.x + threadIdx.x; i < n; i += stride)
        out[i] = f2bf(in[i]);
}

// ---------------------------------------------------------------------------
// Generic bf16 GEMM: C[f32, MxN] = A[bf16, MxK] * W[bf16, KxN]
// 128x128x32 block tile, 256 threads (8 waves, 2x4 wave grid),
// each wave computes a 64x32 tile = 4x2 WMMA C fragments.
// Both LDS tiles are row-major (async-copy friendly); the B fragments are
// assembled with 16-bit gathers along K.
// ---------------------------------------------------------------------------
__global__ __launch_bounds__(256)
void gemm_bf16(const u16* __restrict__ A, const u16* __restrict__ W,
               float* __restrict__ C, int M, int N, int K) {
    __shared__ u16 Xs[128][40];    // [m][k]   row stride 80B  (16B aligned)
    __shared__ u16 Ws[32][136];    // [k][n]   row stride 272B (16B aligned)

    const int t    = threadIdx.x;
    const int lane = t & 31;
    const int w    = t >> 5;
    const int wm   = w >> 2;       // 0..1
    const int wn   = w & 3;        // 0..3
    const int ml   = lane & 15;
    const int hf   = lane >> 4;
    const int m0   = blockIdx.y * 128;
    const int n0   = blockIdx.x * 128;

    v8f acc[4][2];
#pragma unroll
    for (int i = 0; i < 4; ++i)
#pragma unroll
        for (int j = 0; j < 2; ++j) acc[i][j] = v8f_zero();

    for (int kt = 0; kt < K; kt += 32) {
        // ---- stage X tile: 128x32 bf16 = 512 x 16B, 2 chunks/thread
        // ---- stage W tile:  32x128 bf16 = 512 x 16B, 2 chunks/thread
#if ASYNC_COPY
#pragma unroll
        for (int i = 0; i < 2; ++i) {
            int id = t + i * 256;
            int r  = id >> 2, c8 = id & 3;                       // 4 chunks per X row
            g2l_b128(&A[(size_t)(m0 + r) * K + kt + c8 * 8], &Xs[r][c8 * 8]);
        }
#pragma unroll
        for (int i = 0; i < 2; ++i) {
            int id = t + i * 256;
            int r  = id >> 4, c8 = id & 15;                      // 16 chunks per W row
            g2l_b128(&W[(size_t)(kt + r) * N + n0 + c8 * 8], &Ws[r][c8 * 8]);
        }
        async_wait0();
#else
        u32x4 xv[2], wv[2];
#pragma unroll
        for (int i = 0; i < 2; ++i) {
            int id = t + i * 256;
            int r  = id >> 2, c8 = id & 3;
            xv[i] = *(const u32x4*)&A[(size_t)(m0 + r) * K + kt + c8 * 8];
        }
#pragma unroll
        for (int i = 0; i < 2; ++i) {
            int id = t + i * 256;
            int r  = id >> 4, c8 = id & 15;
            wv[i] = *(const u32x4*)&W[(size_t)(kt + r) * N + n0 + c8 * 8];
        }
#pragma unroll
        for (int i = 0; i < 2; ++i) {
            int id = t + i * 256;
            *(u32x4*)&Xs[id >> 2][(id & 3) * 8] = xv[i];
        }
#pragma unroll
        for (int i = 0; i < 2; ++i) {
            int id = t + i * 256;
            *(u32x4*)&Ws[id >> 4][(id & 15) * 8] = wv[i];
        }
#endif
        __syncthreads();

        if (kt + 32 < K) {  // gfx1250 global_prefetch_b8 on next tiles
            __builtin_prefetch(&A[(size_t)(m0 + (t >> 1)) * K + kt + 32], 0, 1);
            __builtin_prefetch(&W[(size_t)(kt + 32 + (t & 31)) * N + n0], 0, 1);
        }

        // ---- A fragments (16x32 bf16): lane m=ml, half hf, packed K pairs
        Frag af[4];
#pragma unroll
        for (int i = 0; i < 4; ++i) {
            int row = wm * 64 + i * 16 + ml;
#pragma unroll
            for (int j = 0; j < 4; ++j) {
                af[i].u[j]     = *(const u32*)&Xs[row][8 * hf + 2 * j];
                af[i].u[4 + j] = *(const u32*)&Xs[row][16 + 8 * hf + 2 * j];
            }
        }
        // ---- B fragments (32x16 bf16): lane n=ml; gather K pairs along rows
        Frag bf[2];
#pragma unroll
        for (int j2 = 0; j2 < 2; ++j2) {
            int col = wn * 32 + j2 * 16 + ml;
#pragma unroll
            for (int j = 0; j < 8; ++j) {
                u32 lo = Ws[8 * hf + 2 * j + 8 * (j >> 2) + 0][col];
                u32 hi = Ws[8 * hf + 2 * j + 8 * (j >> 2) + 1][col];
                bf[j2].u[j] = lo | (hi << 16);
            }
        }
#pragma unroll
        for (int i = 0; i < 4; ++i)
#pragma unroll
            for (int j2 = 0; j2 < 2; ++j2)
                acc[i][j2] = __builtin_amdgcn_wmma_f32_16x16x32_bf16(
                    false, af[i].v, false, bf[j2].v, (short)0, acc[i][j2], false, false);
        __syncthreads();
    }

    // ---- store C (f32): C-layout lane n=ml, row r + 8*hf
#pragma unroll
    for (int i = 0; i < 4; ++i)
#pragma unroll
        for (int j2 = 0; j2 < 2; ++j2)
#pragma unroll
            for (int r = 0; r < 8; ++r) {
                int row = m0 + wm * 64 + i * 16 + r + 8 * hf;
                int col = n0 + wn * 32 + j2 * 16 + ml;
                C[(size_t)row * N + col] = acc[i][j2][r];
            }
}

// ---------------------------------------------------------------------------
// RMSNorm (over HD=64) + RoPE, fp32 in -> bf16 out, head-major [B,NH,L,HD].
// One 64-thread block (2 waves) per (b, l, head).
// ---------------------------------------------------------------------------
__global__ __launch_bounds__(64)
void normrope(const float* __restrict__ in, const float* __restrict__ wgt,
              u16* __restrict__ out, int NH) {
    __shared__ float part[2];
    __shared__ float vals[64];

    int idx = blockIdx.x;
    int h   = idx % NH;
    int l   = (idx / NH) % L_;
    int b   = idx / (NH * L_);
    int d   = threadIdx.x;

    float v  = in[((size_t)(b * L_ + l) * NH + h) * HD_ + d];
    float sq = v * v;
#pragma unroll
    for (int off = 16; off; off >>= 1) sq += __shfl_xor(sq, off, 32);
    if ((d & 31) == 0) part[d >> 5] = sq;
    __syncthreads();
    float var = (part[0] + part[1]) * (1.0f / HD_);
    float tn  = v * rsqrtf(var + 1e-5f) * wgt[d];
    vals[d] = tn;
    __syncthreads();

    int   dh  = d & 31;
    float inv = __powf(10000.0f, -(float)dh * (1.0f / 32.0f));
    float ang = (float)l * inv;
    float s, c;
    __sincosf(ang, &s, &c);
    float x1 = vals[dh], x2 = vals[dh + 32];
    float o  = (d < 32) ? (x1 * c - x2 * s) : (x1 * s + x2 * c);
    out[(((size_t)b * NH + h) * L_ + l) * HD_ + d] = f2bf(o);
}

// ---------------------------------------------------------------------------
// V: fp32 (B*L, HKV*HD) -> bf16 head-major [B,HKV,L,HD]
// ---------------------------------------------------------------------------
__global__ void permute_v(const float* __restrict__ in, u16* __restrict__ out) {
    size_t n = (size_t)B_ * L_ * HKV_ * HD_;
    size_t stride = (size_t)gridDim.x * blockDim.x;
    for (size_t i = (size_t)blockIdx.x * blockDim.x + threadIdx.x; i < n; i += stride) {
        int d = (int)(i % HD_);
        size_t r = i / HD_;
        int h = (int)(r % HKV_); r /= HKV_;
        int l = (int)(r % L_);
        int b = (int)(r / L_);
        out[(((size_t)b * HKV_ + h) * L_ + l) * HD_ + d] = f2bf(in[i]);
    }
}

// ---------------------------------------------------------------------------
// Causal flash attention with fused sigmoid gate.
// grid (L/64, B*H), 128 threads = 4 waves; each wave owns 16 query rows.
// K/V tiles staged row-major via async global->LDS copies.
// S = Q K^T via 2 WMMAs per 16-key halftile, fp32 online softmax,
// P re-layout C->A via per-wave LDS round-trip, O += P V via 4 WMMAs.
// ---------------------------------------------------------------------------
__global__ __launch_bounds__(128)
void flash_attn(const u16* __restrict__ qb, const u16* __restrict__ kb,
                const u16* __restrict__ vb, const float* __restrict__ graw,
                u16* __restrict__ ag) {
    __shared__ u16 Ks[32][72];      // [kv][d], row stride 144B (16B aligned)
    __shared__ u16 Vst[32][72];     // [kv][d], row-major like global
    __shared__ u16 Ps[4][16][32];   // per-wave P tile

    const int t    = threadIdx.x;
    const int w    = t >> 5;
    const int lane = t & 31;
    const int ml   = lane & 15;
    const int hf   = lane >> 4;
    const int qt   = blockIdx.x;
    const int bh   = blockIdx.y;
    const int b    = bh / H_;
    const int h    = bh % H_;
    const int hkv  = h / (H_ / HKV_);
    const int q0   = qt * 64 + w * 16;

    // ---- Q fragments (two: d=0..31 and d=32..63), direct from global
    Frag qf0, qf1;
    {
        size_t base = (((size_t)b * H_ + h) * L_ + (size_t)(q0 + ml)) * HD_;
        const u32* qrow = (const u32*)(qb + base);
#pragma unroll
        for (int j = 0; j < 4; ++j) {
            qf0.u[j]     = qrow[4 * hf + j];           // d = 8*hf + 2j
            qf0.u[4 + j] = qrow[8 + 4 * hf + j];       // d = 16 + 8*hf + 2j
            qf1.u[j]     = qrow[16 + 4 * hf + j];      // d = 32 + 8*hf + 2j
            qf1.u[4 + j] = qrow[24 + 4 * hf + j];      // d = 48 + 8*hf + 2j
        }
    }

    v8f Oc[4];
#pragma unroll
    for (int c = 0; c < 4; ++c) Oc[c] = v8f_zero();
    float mrun[8], lrun[8];
#pragma unroll
    for (int r = 0; r < 8; ++r) { mrun[r] = -1e30f; lrun[r] = 0.0f; }

    const size_t kvbase = (((size_t)b * HKV_ + hkv) * L_) * HD_;
    const int jmax = 2 * qt + 1;

    for (int j = 0; j <= jmax; ++j) {
        const int kv0 = j * 32;
        __syncthreads();
        // ---- stage K and V tiles: each 32x64 bf16 = 256 x 16B, 2 chunks/thread
#if ASYNC_COPY
#pragma unroll
        for (int i = 0; i < 2; ++i) {
            int id = t + i * 128;
            int r  = id >> 3, c8 = id & 7;             // 8 chunks per row
            size_t go = kvbase + (size_t)(kv0 + r) * HD_ + c8 * 8;
            g2l_b128(kb + go, &Ks[r][c8 * 8]);
            g2l_b128(vb + go, &Vst[r][c8 * 8]);
        }
        async_wait0();
#else
        u32x4 kv4[2], vv4[2];
#pragma unroll
        for (int i = 0; i < 2; ++i) {
            int id = t + i * 128;
            int r  = id >> 3, c8 = id & 7;
            size_t go = kvbase + (size_t)(kv0 + r) * HD_ + c8 * 8;
            kv4[i] = *(const u32x4*)(kb + go);
            vv4[i] = *(const u32x4*)(vb + go);
        }
#pragma unroll
        for (int i = 0; i < 2; ++i) {
            int id = t + i * 128;
            int r  = id >> 3, c8 = id & 7;
            *(u32x4*)&Ks[r][c8 * 8]  = kv4[i];
            *(u32x4*)&Vst[r][c8 * 8] = vv4[i];
        }
#endif
        __syncthreads();

        // ---- scores for two 16-key halftiles
        v8f s0v = v8f_zero(), s1v = v8f_zero();
#pragma unroll
        for (int nh = 0; nh < 2; ++nh) {
            int col = nh * 16 + ml;        // key index within 32-wide tile
            Frag bk0, bk1;
#pragma unroll
            for (int j2 = 0; j2 < 8; ++j2) {
                bk0.u[j2] = *(const u32*)&Ks[col][8 * hf + 2 * j2 + 8 * (j2 >> 2)];        // d < 32
                bk1.u[j2] = *(const u32*)&Ks[col][32 + 8 * hf + 2 * j2 + 8 * (j2 >> 2)];   // d >= 32
            }
            v8f sc = v8f_zero();
            sc = __builtin_amdgcn_wmma_f32_16x16x32_bf16(false, qf0.v, false, bk0.v, (short)0, sc, false, false);
            sc = __builtin_amdgcn_wmma_f32_16x16x32_bf16(false, qf1.v, false, bk1.v, (short)0, sc, false, false);
            if (nh == 0) s0v = sc; else s1v = sc;
        }

        // ---- online softmax, rows r+8*hf live across 16 lanes
        const float scale = 0.125f;   // 64^-0.5
#pragma unroll
        for (int r = 0; r < 8; ++r) {
            int qr = q0 + r + 8 * hf;
            float s0 = s0v[r] * scale;
            float s1 = s1v[r] * scale;
            if (kv0 + ml > qr)      s0 = -1e30f;
            if (kv0 + 16 + ml > qr) s1 = -1e30f;
            float tmax = fmaxf(s0, s1);
#pragma unroll
            for (int off = 1; off < 16; off <<= 1)
                tmax = fmaxf(tmax, __shfl_xor(tmax, off, 16));
            float nm    = fmaxf(mrun[r], tmax);
            float alpha = __expf(mrun[r] - nm);
            float p0    = __expf(s0 - nm);
            float p1    = __expf(s1 - nm);
            float rs    = p0 + p1;
#pragma unroll
            for (int off = 1; off < 16; off <<= 1)
                rs += __shfl_xor(rs, off, 16);
            lrun[r] = lrun[r] * alpha + rs;
            mrun[r] = nm;
#pragma unroll
            for (int c = 0; c < 4; ++c) Oc[c][r] *= alpha;
            int m = r + 8 * hf;
            Ps[w][m][ml]      = f2bf(p0);
            Ps[w][m][16 + ml] = f2bf(p1);
        }

        // ---- P: C-layout -> A-layout via per-wave LDS (in-order DS per wave)
        Frag pf;
#pragma unroll
        for (int j2 = 0; j2 < 4; ++j2) {
            pf.u[j2]     = *(const u32*)&Ps[w][ml][8 * hf + 2 * j2];
            pf.u[4 + j2] = *(const u32*)&Ps[w][ml][16 + 8 * hf + 2 * j2];
        }
        // ---- O += P V  (V B-fragments gathered along kv rows)
#pragma unroll
        for (int c = 0; c < 4; ++c) {
            Frag bv;
#pragma unroll
            for (int j2 = 0; j2 < 8; ++j2) {
                u32 lo = Vst[8 * hf + 2 * j2 + 8 * (j2 >> 2) + 0][c * 16 + ml];
                u32 hi = Vst[8 * hf + 2 * j2 + 8 * (j2 >> 2) + 1][c * 16 + ml];
                bv.u[j2] = lo | (hi << 16);
            }
            Oc[c] = __builtin_amdgcn_wmma_f32_16x16x32_bf16(
                false, pf.v, false, bv.v, (short)0, Oc[c], false, false);
        }
    }

    // ---- epilogue: normalize, fuse sigmoid gate, emit bf16 (B*L, H*HD)
#pragma unroll
    for (int c = 0; c < 4; ++c)
#pragma unroll
        for (int r = 0; r < 8; ++r) {
            int qr  = q0 + r + 8 * hf;
            int col = c * 16 + ml;
            float o = Oc[c][r] / lrun[r];
            size_t gi = ((size_t)(b * L_ + qr) * H_ + h) * HD_ + col;
            float g = graw[gi];
            g = 1.0f / (1.0f + __expf(-g));
            ag[gi] = f2bf(o * g);
        }
}

// ---------------------------------------------------------------------------
// Host-side orchestration
// ---------------------------------------------------------------------------
extern "C" void kernel_launch(void* const* d_in, const int* in_sizes, int n_in,
                              void* d_out, int out_size, void* d_ws, size_t ws_size,
                              hipStream_t stream) {
    const float* x  = (const float*)d_in[0];
    const float* Wq = (const float*)d_in[1];
    const float* Wk = (const float*)d_in[2];
    const float* Wv = (const float*)d_in[3];
    const float* Wg = (const float*)d_in[4];
    const float* Wo = (const float*)d_in[5];
    const float* qn = (const float*)d_in[6];
    const float* kn = (const float*)d_in[7];
    // d_in[8] = causal mask: known analytically, unused.

    size_t off = 0;
    auto alloc = [&](size_t bytes) -> void* {
        void* p = (char*)d_ws + off;
        off += (bytes + 255) & ~(size_t)255;
        return p;
    };

    u16*   xb   = (u16*)alloc((size_t)M_ * D_ * 2);
    u16*   wqb  = (u16*)alloc((size_t)D_ * NQ_ * 2);
    u16*   wkb  = (u16*)alloc((size_t)D_ * NKV_ * 2);
    u16*   wvb  = (u16*)alloc((size_t)D_ * NKV_ * 2);
    u16*   wgb  = (u16*)alloc((size_t)D_ * NQ_ * 2);
    u16*   wob  = (u16*)alloc((size_t)NQ_ * D_ * 2);
    float* qraw = (float*)alloc((size_t)M_ * NQ_ * 4);
    float* kraw = (float*)alloc((size_t)M_ * NKV_ * 4);
    float* vraw = (float*)alloc((size_t)M_ * NKV_ * 4);
    float* graw = (float*)alloc((size_t)M_ * NQ_ * 4);
    u16*   qbf  = (u16*)alloc((size_t)M_ * NQ_ * 2);
    u16*   kbf  = (u16*)alloc((size_t)M_ * NKV_ * 2);
    u16*   vbf  = (u16*)alloc((size_t)M_ * NKV_ * 2);
    u16*   ag   = (u16*)alloc((size_t)M_ * NQ_ * 2);

    // 1) bf16 conversions
    cvt_f32_bf16<<<2048, 256, 0, stream>>>(x,  xb,  (size_t)M_ * D_);
    cvt_f32_bf16<<<2048, 256, 0, stream>>>(Wq, wqb, (size_t)D_ * NQ_);
    cvt_f32_bf16<<<512,  256, 0, stream>>>(Wk, wkb, (size_t)D_ * NKV_);
    cvt_f32_bf16<<<512,  256, 0, stream>>>(Wv, wvb, (size_t)D_ * NKV_);
    cvt_f32_bf16<<<2048, 256, 0, stream>>>(Wg, wgb, (size_t)D_ * NQ_);
    cvt_f32_bf16<<<2048, 256, 0, stream>>>(Wo, wob, (size_t)NQ_ * D_);

    // 2) projections (WMMA GEMMs)
    gemm_bf16<<<dim3(NQ_ / 128,  M_ / 128), 256, 0, stream>>>(xb, wqb, qraw, M_, NQ_,  D_);
    gemm_bf16<<<dim3(NKV_ / 128, M_ / 128), 256, 0, stream>>>(xb, wkb, kraw, M_, NKV_, D_);
    gemm_bf16<<<dim3(NKV_ / 128, M_ / 128), 256, 0, stream>>>(xb, wvb, vraw, M_, NKV_, D_);
    gemm_bf16<<<dim3(NQ_ / 128,  M_ / 128), 256, 0, stream>>>(xb, wgb, graw, M_, NQ_,  D_);

    // 3) RMSNorm + RoPE -> head-major bf16; V permute/convert
    normrope<<<(size_t)B_ * L_ * H_,   64, 0, stream>>>(qraw, qn, qbf, H_);
    normrope<<<(size_t)B_ * L_ * HKV_, 64, 0, stream>>>(kraw, kn, kbf, HKV_);
    permute_v<<<1024, 256, 0, stream>>>(vraw, vbf);

    // 4) causal flash attention + fused sigmoid gate
    flash_attn<<<dim3(L_ / 64, B_ * H_), 128, 0, stream>>>(qbf, kbf, vbf, graw, ag);

    // 5) output projection -> fp32 d_out
    gemm_bf16<<<dim3(D_ / 128, M_ / 128), 256, 0, stream>>>(ag, wob, (float*)d_out, M_, D_, NQ_);
}
